// FF_30545807409995
// MI455X (gfx1250) — compile-verified, matrix operand order checked by reading, not run on hardware
//
#include <hip/hip_runtime.h>

// ---------------------------------------------------------------------------
// Problem: B=16, L=512, D_MODEL=1024, D_FF=4096
//   s = cumsum(x, L);  r[b,i] = sum_e relu(s[b,i,:] @ W0[:,e]);  t = cumsum(r);
//   out[b,i,d] = t[b,i] * W1[d,i]
// Hot path: bf16 WMMA (v_wmma_f32_16x16x32_bf16), f32 accumulation.
// B (W0) slabs are DMA'd into LDS with the Tensor Data Mover (double-buffered,
// tensor_load_to_lds + s_wait_tensorcnt) and shared by all 8 waves -> 8x less
// L2 traffic than per-wave global loads. A streams from global (coalesced 32B
// per lane, fragment-order pre-swizzled).
// ---------------------------------------------------------------------------

typedef __attribute__((ext_vector_type(16))) __bf16        v16bf;
typedef __attribute__((ext_vector_type(8)))  float         v8f;
typedef __attribute__((ext_vector_type(4)))  unsigned int  u32x4;
typedef __attribute__((ext_vector_type(8)))  int           i32x8;
typedef __attribute__((ext_vector_type(4)))  int           i32x4;

#define B_  16
#define L_  512
#define DM  1024
#define DF  4096

static __device__ __forceinline__ unsigned short f32_to_bf16(float f) {
  union { float f; unsigned u; } v; v.f = f;
  unsigned r = v.u + 0x7FFFu + ((v.u >> 16) & 1u);   // round to nearest even
  return (unsigned short)(r >> 16);
}

// ---------------------------------------------------------------------------
// TDM: DMA a B slab (8 fragment-rows x 4KB, row stride 32KB) into LDS.
// D# per CDNA5 ISA: 2-D tensor, data_size=1B,
// tile_dim0=4096, tile_dim1=8, tensor_dim0_stride=32768, type=2.
// ---------------------------------------------------------------------------
static __device__ __forceinline__ void tdm_load_slab(const void* gsrc, unsigned lds_off) {
  const unsigned long long ga = (unsigned long long)gsrc;
  u32x4 g0;
  g0[0] = 1u;                                               // count=1, user mode
  g0[1] = lds_off;                                          // lds_addr (bytes)
  g0[2] = (unsigned)(ga & 0xFFFFFFFFull);                   // global_addr[31:0]
  g0[3] = (unsigned)((ga >> 32) & 0x1FFFFFFull)             // global_addr[56:32]
          | 0x80000000u;                                    // type=2 (bits 127:126)
  i32x8 g1;
  g1[0] = 0;                                                // wg_mask=0, data_size=1B
  g1[1] = (int)0x80000000u;                                 // tensor_dim0=32768 (bits 63:48)
  g1[2] = (int)(4096u << 16);                               // tensor_dim1=4096 (bits 95:80)
  g1[3] = (int)(4096u << 16);                               // tile_dim0=4096 (bits 127:112)
  g1[4] = 8;                                                // tile_dim1=8 (bits 143:128)
  g1[5] = 32768;                                            // tensor_dim0_stride[31:0]
  g1[6] = 0;
  g1[7] = 0;
  i32x4 z4 = {0, 0, 0, 0};
#if __clang_major__ >= 23
  i32x8 z8 = {0, 0, 0, 0, 0, 0, 0, 0};
  __builtin_amdgcn_tensor_load_to_lds(g0, g1, z4, z4, z8, 0);
#else
  __builtin_amdgcn_tensor_load_to_lds(g0, g1, z4, z4, 0);
#endif
}

// ---------------------------------------------------------------------------
// K1: W0 [K=1024, N=4096] f32 -> fragment-order bf16 (16-bit B layout:
// lane l holds col n = nt*16+(l&15); halves 0..7 = K kt*32+(l>>4)*8+h,
// halves 8..15 = K kt*32+16+(l>>4)*8+(h-8)).
// ---------------------------------------------------------------------------
__global__ void prep_w0(const float* __restrict__ w0, unsigned short* __restrict__ w0f) {
  const int blk  = blockIdx.x;           // 0 .. 256*32-1  (nt, kt)
  const int nt   = blk >> 5;
  const int kt   = blk & 31;
  const int lane = threadIdx.x;          // 32 threads
  const int n    = nt * 16 + (lane & 15);
  const int lh   = lane >> 4;
  unsigned short* dst = w0f + (size_t)blk * 512 + lane * 16;
  #pragma unroll
  for (int h = 0; h < 16; ++h) {
    const int koff = (h < 8) ? (lh * 8 + h) : (16 + lh * 8 + (h - 8));
    dst[h] = f32_to_bf16(w0[(size_t)(kt * 32 + koff) * DF + n]);
  }
}

// ---------------------------------------------------------------------------
// K2: s = cumsum(x over L), written in A-fragment order (16-bit A layout).
// ---------------------------------------------------------------------------
__global__ void cumsum_swizzle(const float* __restrict__ x, unsigned short* __restrict__ sA) {
  const int t = blockIdx.x * 256 + threadIdx.x;      // < 16384
  const int b = t >> 10;
  const int d = t & (DM - 1);
  const float* xp = x + (size_t)b * L_ * DM + d;
  const int kt   = d >> 5;
  const int kd   = d & 31;
  const int lane = ((kd >> 3) & 1) ? 16 : 0;
  const int h    = (kd & 7) + ((kd >> 4) << 3);
  float run = 0.0f;
  for (int l = 0; l < L_; ++l) {
    run += xp[(size_t)l * DM];
    const int mt = l >> 4, m = l & 15;
    const size_t idx = (((size_t)(b * 32 + mt) * 32) + kt) * 512 + (lane + m) * 16 + h;
    sA[idx] = f32_to_bf16(run);
  }
}

__global__ void zero_f(float* __restrict__ p, int n) {
  const int i = blockIdx.x * 256 + threadIdx.x;
  if (i < n) p[i] = 0.0f;
}

// ---------------------------------------------------------------------------
// K4: fused GEMM + relu + row-sum.
// Grid (8 N-blocks, 4 M-blocks, 16 batches) x 256 threads (8 waves).
// Wave w owns one 16-row M-tile; WG covers 512 cols as 4 chunks of 8 N-tiles.
// Per chunk: 8 live f32 accumulator tiles; K=1024 as 8 slabs of 4 K-steps.
// B slabs (32KB = 8 nt x 4 kt fragments) are TDM-DMA'd into LDS, double
// buffered: wave0 issues tensor_load_to_lds for slab s+1, all waves compute
// slab s from LDS (2x ds_load_b128 per fragment), wave0 s_wait_tensorcnt,
// barrier, swap.
// ---------------------------------------------------------------------------
__global__ void __launch_bounds__(256)
gemm_relu_rowsum(const unsigned short* __restrict__ sA,
                 const unsigned short* __restrict__ w0f,
                 float* __restrict__ rsum) {
  __shared__ uint4 ldsB[2][2048];                    // 2 x 32KB slabs

  const unsigned lane  = threadIdx.x & 31u;
  const unsigned wave  = threadIdx.x >> 5;
  const unsigned b     = blockIdx.z;
  const unsigned mtile = blockIdx.y * 8 + wave;      // 0..31

  const v16bf* Afr = ((const v16bf*)sA) + ((size_t)(b * 32 + mtile) * 32) * 32 + lane;

  float rowsum[8];
  #pragma unroll
  for (int r = 0; r < 8; ++r) rowsum[r] = 0.0f;

  for (int chunk = 0; chunk < 4; ++chunk) {
    const unsigned nt0 = blockIdx.x * 32 + chunk * 8;
    // fragment (nt0+j, kt) lives at ((nt0+j)*32 + kt)*512 ushorts; slab row
    // stride = 32KB, slab (ktBase) starts at (nt0*32 + ktBase)*512.
    const unsigned short* slab0 = w0f + (size_t)nt0 * 32 * 512;

    v8f acc[8];
    #pragma unroll
    for (int j = 0; j < 8; ++j)
      #pragma unroll
      for (int r = 0; r < 8; ++r) acc[j][r] = 0.0f;

    // preload slab 0 into buffer 0
    if (wave == 0) {
      tdm_load_slab(slab0, (unsigned)(size_t)(void*)&ldsB[0][0]);
      __builtin_amdgcn_s_wait_tensorcnt(0);
    }
    __syncthreads();

    unsigned cur = 0;
    for (int s = 0; s < 8; ++s) {
      if (wave == 0 && s + 1 < 8)                     // DMA next slab, overlapped
        tdm_load_slab(slab0 + (size_t)(s + 1) * 4 * 512,
                      (unsigned)(size_t)(void*)&ldsB[cur ^ 1][0]);

      const int ktBase = s * 4;
      #pragma unroll
      for (int kl = 0; kl < 4; ++kl) {
        const v16bf a = Afr[(size_t)(ktBase + kl) * 32];      // 32B coalesced
        #pragma unroll
        for (int j = 0; j < 8; ++j) {
          // LDS slab layout: [nt j][kt kl][1KB fragment], lane-major inside
          const v16bf bb = *(const v16bf*)&ldsB[cur][(j * 4 + kl) * 64 + lane * 2];
          acc[j] = __builtin_amdgcn_wmma_f32_16x16x32_bf16(
              false, a, false, bb, (short)0, acc[j], false, false);
        }
      }

      if (wave == 0) __builtin_amdgcn_s_wait_tensorcnt(0);
      __syncthreads();
      cur ^= 1;
    }

    // relu + per-row partial sums (row = r + 8*(lane>>4), col = lane&15)
    #pragma unroll
    for (int j = 0; j < 8; ++j)
      #pragma unroll
      for (int r = 0; r < 8; ++r)
        rowsum[r] += fmaxf(acc[j][r], 0.0f);
  }

  // reduce across the 16 N-lanes of each half-wave
  #pragma unroll
  for (int r = 0; r < 8; ++r) {
    float v = rowsum[r];
    v += __shfl_xor(v, 1);
    v += __shfl_xor(v, 2);
    v += __shfl_xor(v, 4);
    v += __shfl_xor(v, 8);
    if ((lane & 15u) == 0u) {
      const unsigned row = mtile * 16 + (lane >> 4) * 8 + r;
      atomicAdd(&rsum[b * L_ + row], v);
    }
  }
}

// ---------------------------------------------------------------------------
__global__ void prefix_r(float* __restrict__ r) {
  const int b = threadIdx.x;
  if (b < B_) {
    float run = 0.0f;
    for (int i = 0; i < L_; ++i) { run += r[b * L_ + i]; r[b * L_ + i] = run; }
  }
}

__global__ void transpose_w1(const float* __restrict__ w1, float* __restrict__ w1t) {
  __shared__ float tile[32][33];
  const int i0 = blockIdx.x * 32;
  const int d0 = blockIdx.y * 32;
  const int tx = threadIdx.x, ty = threadIdx.y;      // 32 x 8
  #pragma unroll
  for (int j = 0; j < 32; j += 8)
    tile[ty + j][tx] = w1[(size_t)(d0 + ty + j) * L_ + (i0 + tx)];
  __syncthreads();
  #pragma unroll
  for (int j = 0; j < 32; j += 8)
    w1t[(size_t)(i0 + ty + j) * DM + (d0 + tx)] = tile[tx][ty + j];
}

__global__ void finalize(const float* __restrict__ t, const float* __restrict__ w1t,
                         float* __restrict__ out) {
  const size_t idx = (size_t)blockIdx.x * 256 + threadIdx.x;  // < 8388608
  const int b = (int)(idx >> 19);
  const int i = (int)((idx >> 10) & (L_ - 1));
  const int d = (int)(idx & (DM - 1));
  out[idx] = t[b * L_ + i] * w1t[(size_t)i * DM + d];
}

// ---------------------------------------------------------------------------
extern "C" void kernel_launch(void* const* d_in, const int* in_sizes, int n_in,
                              void* d_out, int out_size, void* d_ws, size_t ws_size,
                              hipStream_t stream) {
  const float* x  = (const float*)d_in[0];   // [16,512,1024]
  const float* w0 = (const float*)d_in[1];   // [1024,4096]
  const float* w1 = (const float*)d_in[2];   // [1024,512]
  float* out = (float*)d_out;

  char* ws = (char*)d_ws;
  unsigned short* sA   = (unsigned short*)(ws + 0);            // 16 MB
  unsigned short* w0f  = (unsigned short*)(ws + (16u << 20));  //  8 MB
  float*          w1t  = (float*)(ws + (24u << 20));           //  2 MB
  float*          rbuf = (float*)(ws + (26u << 20));           // 32 KB

  prep_w0       <<<dim3(256 * 32), dim3(32),    0, stream>>>(w0, w0f);
  transpose_w1  <<<dim3(16, 32),   dim3(32, 8), 0, stream>>>(w1, w1t);
  cumsum_swizzle<<<dim3(64),       dim3(256),   0, stream>>>(x, sA);
  zero_f        <<<dim3(32),       dim3(256),   0, stream>>>(rbuf, B_ * L_);

  gemm_relu_rowsum<<<dim3(8, 4, 16), dim3(256), 0, stream>>>(sA, w0f, rbuf);

  prefix_r<<<dim3(1),     dim3(16),  0, stream>>>(rbuf);
  finalize<<<dim3(32768), dim3(256), 0, stream>>>(rbuf, w1t, out);
}
